// SwinTransformer_82841329206092
// MI455X (gfx1250) — matrix operand order repaired and checked by reading, not compiled, not run on hardware
//
#include <hip/hip_runtime.h>
#include <hip/hip_bf16.h>

// ------------------------------------------------------------------
// CDNA5 (gfx1250) Swin-Transformer forward.
// wave32, bf16-in-HBM GEMM operands, f32 WMMA accumulation,
// TDM (tensor_load_to_lds) double-buffered GEMM staging when available.
// ------------------------------------------------------------------

typedef __attribute__((ext_vector_type(16))) __bf16 v16bf;
typedef __attribute__((ext_vector_type(8)))  __bf16 v8bf;
typedef __attribute__((ext_vector_type(8)))  float  v8f;
typedef __attribute__((ext_vector_type(4)))  unsigned int v4u;
typedef __attribute__((ext_vector_type(8)))  int v8i;
typedef __attribute__((ext_vector_type(4)))  int v4i;

__device__ __forceinline__ __bf16 f2bf(float f) { return (__bf16)f; }

#define WMMA_BF16(A_, B_, C_) \
  __builtin_amdgcn_wmma_f32_16x16x32_bf16(false, (A_), false, (B_), (short)0, (C_), false, false)

#if defined(__has_builtin)
#if __has_builtin(__builtin_amdgcn_tensor_load_to_lds) && \
    __has_builtin(__builtin_amdgcn_s_wait_tensorcnt)
#define HAVE_TDM 1
#endif
#endif
#ifndef HAVE_TDM
#define HAVE_TDM 0
#endif

#if HAVE_TDM
// Generic LDS pointer low 32 bits == LDS byte address
__device__ __forceinline__ unsigned lds_off32(const void* p) {
  return (unsigned)(unsigned long long)p;
}
// TDM 2-D bf16 tile load, tile_dim0 = 32 elems (64B rows) into LDS rows padded
// to 80B (pad_interval = 16 dwords -> code 3, pad_amount = 4 dwords -> code 3).
// tensor_dim0/1 = remaining cols/rows so out-of-tensor reads return zero
// (hardware handles ragged K / ragged M,N).
// NOTE: this toolchain exposes the 6-arg builtin
//   (uint32x4 g0, int32x8 g1, int32x4 g2, int32x4 g3, int32x8 g4, i32 cpol)
__device__ __forceinline__ void tdm_load_tile(const __bf16* gsrc, unsigned lds_byte,
                                              int tile_rows, int rows_rem,
                                              int cols_rem, int row_stride_elems)
{
  unsigned long long ga = (unsigned long long)gsrc;
  unsigned td0 = (unsigned)cols_rem;      // tensor_dim0 (elems)
  unsigned td1 = (unsigned)rows_rem;      // tensor_dim1 (rows)
  v4u g0;
  g0[0] = 1u;                                               // count=1, user mode
  g0[1] = lds_byte;                                         // lds_addr
  g0[2] = (unsigned)(ga & 0xFFFFFFFFull);                   // global_addr[31:0]
  g0[3] = (unsigned)((ga >> 32) & 0x1FFFFFFull) | (2u << 30); // addr[56:32], type=2
  v8i g1;
  g1[0] = (int)((1u << 16) | (1u << 20) | (3u << 22) | (3u << 25)); // data_size=2B, pad on
  g1[1] = (int)((td0 & 0xFFFFu) << 16);                     // barrier_addr=0 | td0 lo
  g1[2] = (int)((td0 >> 16) | ((td1 & 0xFFFFu) << 16));     // td0 hi | td1 lo
  g1[3] = (int)((td1 >> 16) | (32u << 16));                 // td1 hi | tile_dim0=32
  g1[4] = (int)((unsigned)tile_rows & 0xFFFFu);             // tile_dim1 | tile_dim2=0
  g1[5] = (int)(unsigned)row_stride_elems;                  // tensor_dim0_stride[31:0]
  g1[6] = 0;                                                // stride0 hi | stride1 lo
  g1[7] = 0;
  v4i g2; g2[0] = 0; g2[1] = 0; g2[2] = 0; g2[3] = 0;       // 2-D: dims 2/3 unused
  v4i g3; g3[0] = 0; g3[1] = 0; g3[2] = 0; g3[3] = 0;
  v8i g4; g4[0] = 0; g4[1] = 0; g4[2] = 0; g4[3] = 0;
          g4[4] = 0; g4[5] = 0; g4[6] = 0; g4[7] = 0;
  __builtin_amdgcn_tensor_load_to_lds(g0, g1, g2, g3, g4, 0);
}
#endif

// ==================================================================
// GEMM: C[M,N] = epilogue(A[M,K] @ B[K,N]),  B pre-transposed BT[N,K], bf16.
// 256 threads = 8 waves (4x2), tile 128x64, BK=32, double-buffered LDS.
// TDM path: waves 0/1 DMA next k-tile while all waves run WMMA on current.
// epilogue: (+bias) -> (silu?) -> (+resid); f32 or bf16 output.
// ==================================================================
__global__ __launch_bounds__(256)
void gemm_bf16_kernel(const __bf16* __restrict__ A, const __bf16* __restrict__ BT,
                      const float* __restrict__ bias, const float* __restrict__ resid,
                      void* __restrict__ Cout, int M, int N, int K,
                      int act_silu, int out_bf16)
{
  __shared__ __bf16 As[2][128][40];   // [buf][m][k], 80B row stride
  __shared__ __bf16 Bs[2][64][40];    // [buf][n][k] (transposed storage)

  const int tid  = threadIdx.x;
  const int wave = tid >> 5;
  const int lane = tid & 31;
  const int hl   = lane >> 4;
  const int lr   = lane & 15;
  const int wm   = wave >> 1;         // 0..3 -> 32 rows
  const int wn   = wave & 1;          // 0..1 -> 32 cols
  const int m0   = blockIdx.x * 128;
  const int n0   = blockIdx.y * 64;

  v8f acc[2][2] = {};
  const int nk = (K + 31) >> 5;

#if HAVE_TDM
  if (wave == 0)
    tdm_load_tile(A + (size_t)m0 * K, lds_off32(&As[0][0][0]), 128, M - m0, K, K);
  if (wave == 1)
    tdm_load_tile(BT + (size_t)n0 * K, lds_off32(&Bs[0][0][0]), 64, N - n0, K, K);
#endif

  for (int kt = 0; kt < nk; ++kt) {
    const int kb  = kt << 5;
    const int cur = kt & 1;
#if HAVE_TDM
    (void)kb;
    if (kt + 1 < nk) {
      const int kn = (kt + 1) << 5;
      if (wave == 0)
        tdm_load_tile(A + (size_t)m0 * K + kn, lds_off32(&As[cur ^ 1][0][0]),
                      128, M - m0, K - kn, K);
      if (wave == 1)
        tdm_load_tile(BT + (size_t)n0 * K + kn, lds_off32(&Bs[cur ^ 1][0][0]),
                      64, N - n0, K - kn, K);
      if (wave < 2) __builtin_amdgcn_s_wait_tensorcnt((short)1);
    } else {
      if (wave < 2) __builtin_amdgcn_s_wait_tensorcnt((short)0);
    }
#else
    // vector-load staging fallback (16B chunks)
    for (int i = tid; i < 512; i += 256) {
      int r = i >> 2, k8 = (i & 3) << 3;
      int gm = m0 + r, gk = kb + k8;
      v8bf val = {};
      if (gm < M && gk < K) val = *(const v8bf*)(A + (size_t)gm * K + gk);
      *(v8bf*)(&As[cur][r][k8]) = val;
    }
    {
      int r = tid >> 2, k8 = (tid & 3) << 3;
      int gn = n0 + r, gk = kb + k8;
      v8bf val = {};
      if (gn < N && gk < K) val = *(const v8bf*)(BT + (size_t)gn * K + gk);
      *(v8bf*)(&Bs[cur][r][k8]) = val;
    }
#endif
    __syncthreads();

    // A frags: per lane one contiguous 16-elem chunk -> 2x ds_load_b128
    v16bf af[2], bfg[2];
#pragma unroll
    for (int i = 0; i < 2; ++i) {
      const __bf16* ap = &As[cur][wm * 32 + i * 16 + lr][0];
#pragma unroll
      for (int v = 0; v < 8; ++v) {
        int kk = (hl << 4) + (v << 1);
        af[i][2 * v]     = ap[kk];
        af[i][2 * v + 1] = ap[kk + 1];
      }
    }
    // B frags: per lane one contiguous 16-elem chunk -> 2x ds_load_b128
#pragma unroll
    for (int j = 0; j < 2; ++j) {
      const __bf16* bp = &Bs[cur][wn * 32 + j * 16 + lr][hl << 4];
#pragma unroll
      for (int e = 0; e < 16; ++e) bfg[j][e] = bp[e];
    }
    acc[0][0] = WMMA_BF16(af[0], bfg[0], acc[0][0]);
    acc[0][1] = WMMA_BF16(af[0], bfg[1], acc[0][1]);
    acc[1][0] = WMMA_BF16(af[1], bfg[0], acc[1][0]);
    acc[1][1] = WMMA_BF16(af[1], bfg[1], acc[1][1]);
    __syncthreads();
  }

  // epilogue (C layout: lane%16 -> N, vgpr r + 8*(lane/16) -> M)
#pragma unroll
  for (int i = 0; i < 2; ++i)
#pragma unroll
    for (int j = 0; j < 2; ++j) {
      int col = n0 + wn * 32 + j * 16 + lr;
      if (col >= N) continue;
      float bv = bias ? bias[col] : 0.f;
#pragma unroll
      for (int r = 0; r < 8; ++r) {
        int row = m0 + wm * 32 + i * 16 + r + (hl << 3);
        if (row >= M) continue;
        float v = acc[i][j][r] + bv;
        if (act_silu) v = v / (1.f + __expf(-v));
        if (resid) v += resid[(size_t)row * N + col];
        if (out_bf16) ((__bf16*)Cout)[(size_t)row * N + col] = f2bf(v);
        else          ((float*) Cout)[(size_t)row * N + col] = v;
      }
    }
}

// ==================================================================
// Windowed attention, one block (128 thr, 4 waves) per (image, head, window).
// qkv bf16 [B,H,W,3*nh*32] -> out bf16 [B,H,W,nh*32].  T=49 padded to 64.
// ==================================================================
__global__ __launch_bounds__(128)
void win_attn_kernel(const __bf16* __restrict__ qkv, const float* __restrict__ relpos,
                     __bf16* __restrict__ out, int Himg, int Wimg, int nh, int shifted)
{
  const int ncols = Wimg / 7, nrows = Himg / 7, nwin = nrows * ncols;
  int bid = blockIdx.x;
  const int widx = bid % nwin;  bid /= nwin;
  const int head = bid % nh;    const int b = bid / nh;
  const int wi = widx / ncols, wj = widx % ncols;
  const int CH = 3 * nh * 32;

  __shared__ __bf16 Qs[64][40];    // [t][d]
  __shared__ __bf16 Ks[64][40];    // [t][d]
  __shared__ __bf16 VsT[32][72];   // [d][t] transposed
  __shared__ float  Ss[64][68];
  __shared__ __bf16 Ps[64][72];

  const int tid = threadIdx.x;
  for (int i = tid; i < 64 * 4; i += 128) {
    int t = i >> 2, d8 = (i & 3) << 3;
    v8bf q = {}, k = {}, vv = {};
    if (t < 49) {
      int h = wi * 7 + t / 7, w = wj * 7 + t % 7;
      const __bf16* p = qkv + ((size_t)(b * Himg + h) * Wimg + w) * CH + head * 32 + d8;
      q  = *(const v8bf*)(p);
      k  = *(const v8bf*)(p + nh * 32);
      vv = *(const v8bf*)(p + 2 * nh * 32);
    }
    *(v8bf*)(&Qs[t][d8]) = q;
    *(v8bf*)(&Ks[t][d8]) = k;
#pragma unroll
    for (int e = 0; e < 8; ++e) VsT[d8 + e][t] = vv[e];
  }
  __syncthreads();

  const int wave = tid >> 5, lane = tid & 31, hl = lane >> 4, lr = lane & 15;

  // ---- S = (Q K^T)/sqrt(dh) + rel_pos_bias (+ shift mask)
  v16bf aq;
  {
    const __bf16* ap = &Qs[wave * 16 + lr][0];
#pragma unroll
    for (int v = 0; v < 8; ++v) {
      int kk = (hl << 4) + (v << 1);
      aq[2 * v] = ap[kk]; aq[2 * v + 1] = ap[kk + 1];
    }
  }
  const bool maskwin = shifted && (wi == nrows - 1);  // UL mask; LR mask all-False (source bug kept)
  for (int ct = 0; ct < 4; ++ct) {
    int u = ct * 16 + lr;
    v16bf bk;
    const __bf16* kp = &Ks[u][hl << 4];
#pragma unroll
    for (int e = 0; e < 16; ++e) bk[e] = kp[e];
    v8f s = {};
    s = WMMA_BF16(aq, bk, s);
#pragma unroll
    for (int r = 0; r < 8; ++r) {
      int t = wave * 16 + r + (hl << 3);
      float val;
      if (u >= 49) {
        val = -1e30f;
      } else if (t >= 49) {
        val = 0.f;
      } else {
        int ti = t / 7, tj = t % 7, ui = u / 7, uj = u % 7;
        int pos = (ui - ti + 6) * 13 + (uj - tj + 6);
        val = s[r] * 0.17677669529663689f + relpos[(size_t)pos * nh + head];
        if (maskwin && ((t >= 28) != (u >= 28))) val = -1e30f;
      }
      Ss[t][u] = val;
    }
  }
  __syncthreads();

  if (tid < 64) {
    float m = -1e30f;
    for (int u = 0; u < 64; ++u) m = fmaxf(m, Ss[tid][u]);
    float sum = 0.f;
    for (int u = 0; u < 64; ++u) sum += __expf(Ss[tid][u] - m);
    float inv = 1.f / sum;
    for (int u = 0; u < 64; ++u) Ps[tid][u] = f2bf(__expf(Ss[tid][u] - m) * inv);
  }
  __syncthreads();

  // ---- O = P @ V
#pragma unroll
  for (int nt = 0; nt < 2; ++nt) {
    v8f o = {};
#pragma unroll
    for (int ks = 0; ks < 2; ++ks) {
      v16bf ap, bv;
      const __bf16* pr = &Ps[wave * 16 + lr][ks * 32];
#pragma unroll
      for (int v = 0; v < 8; ++v) {
        int kk = (hl << 4) + (v << 1);
        ap[2 * v] = pr[kk]; ap[2 * v + 1] = pr[kk + 1];
      }
      const __bf16* vp = &VsT[nt * 16 + lr][ks * 32 + (hl << 4)];
#pragma unroll
      for (int e = 0; e < 16; ++e) bv[e] = vp[e];
      o = WMMA_BF16(ap, bv, o);
    }
    int d = nt * 16 + lr;
#pragma unroll
    for (int r = 0; r < 8; ++r) {
      int t = wave * 16 + r + (hl << 3);
      if (t < 49) {
        int h = wi * 7 + t / 7, w = wj * 7 + t % 7;
        out[((size_t)(b * Himg + h) * Wimg + w) * (nh * 32) + head * 32 + d] = f2bf(o[r]);
      }
    }
  }
}

// ==================================================================
// LayerNorm (f32 in, f32 or bf16 out): one wave per row
// ==================================================================
template <typename OUT>
__global__ __launch_bounds__(256)
void layernorm_kernel(const float* __restrict__ x, const float* __restrict__ g,
                      const float* __restrict__ b, OUT* __restrict__ y,
                      int M, int D, float eps)
{
  int row  = blockIdx.x * (blockDim.x >> 5) + (threadIdx.x >> 5);
  int lane = threadIdx.x & 31;
  if (row >= M) return;
  const float* xr = x + (size_t)row * D;
  float s = 0.f, ss = 0.f;
  for (int i = lane; i < D; i += 32) { float v = xr[i]; s += v; ss += v * v; }
#pragma unroll
  for (int o = 16; o > 0; o >>= 1) { s += __shfl_xor(s, o, 32); ss += __shfl_xor(ss, o, 32); }
  float mu  = s / D;
  float var = ss / D - mu * mu;
  float inv = rsqrtf(var + eps);
  OUT* yr = y + (size_t)row * D;
  for (int i = lane; i < D; i += 32) yr[i] = (OUT)((xr[i] - mu) * inv * g[i] + b[i]);
}

// ==================================================================
// Patch embed gather -> bf16 A matrix [B*Ho*Wo, s*s*C], inner (si,sj,c)
// ==================================================================
__global__ __launch_bounds__(256)
void patch_gather_kernel(const float* __restrict__ src, __bf16* __restrict__ dst,
                         int B, int Hin, int Win, int Cin, int s, int nchw)
{
  int Ho = Hin / s, Wo = Win / s;
  size_t total = (size_t)B * Ho * Wo * s * s * Cin;
  size_t i = (size_t)blockIdx.x * blockDim.x + threadIdx.x;
  if (i >= total) return;
  size_t t = i;
  int c  = t % Cin; t /= Cin;
  int sj = t % s;   t /= s;
  int si = t % s;   t /= s;
  int wo = t % Wo;  t /= Wo;
  int ho = t % Ho;
  int b  = t / Ho;
  int h = ho * s + si, w = wo * s + sj;
  float v = nchw ? src[(((size_t)b * Cin + c) * Hin + h) * Win + w]
                 : src[(((size_t)b * Hin + h) * Win + w) * Cin + c];
  dst[i] = f2bf(v);
}

// Weight convert + transpose: W f32 [K,N] -> WT bf16 [N,K]
__global__ __launch_bounds__(256)
void transpose_w_kernel(const float* __restrict__ W, __bf16* __restrict__ WT, int K, int N)
{
  size_t i = (size_t)blockIdx.x * blockDim.x + threadIdx.x;
  if (i >= (size_t)K * N) return;
  size_t n = i / K, k = i % K;
  WT[i] = f2bf(W[k * N + n]);
}

// bf16 roll (jnp.roll: dst[i] = src[(i - sh) mod n]), vectorized by 8 channels
__global__ __launch_bounds__(256)
void roll_bf16_kernel(const __bf16* __restrict__ src, __bf16* __restrict__ dst,
                      int B, int H, int W, int C, int sh, int sw)
{
  int C8 = C >> 3;
  size_t total = (size_t)B * H * W * C8;
  size_t i = (size_t)blockIdx.x * blockDim.x + threadIdx.x;
  if (i >= total) return;
  size_t t = i;
  int c8 = t % C8; t /= C8;
  int w  = t % W;  t /= W;
  int h  = t % H;
  int b  = t / H;
  int hh = (h - sh) % H; if (hh < 0) hh += H;
  int ww = (w - sw) % W; if (ww < 0) ww += W;
  *(v8bf*)(dst + i * 8) =
      *(const v8bf*)(src + ((((size_t)b * H + hh) * W + ww) * C8 + c8) * 8);
}

// f32 residual accumulate with roll-back: x += roll(src, sh, sw)
__global__ __launch_bounds__(256)
void roll_add_kernel(float* __restrict__ x, const float* __restrict__ src,
                     int B, int H, int W, int C, int sh, int sw)
{
  size_t total = (size_t)B * H * W * C;
  size_t i = (size_t)blockIdx.x * blockDim.x + threadIdx.x;
  if (i >= total) return;
  size_t t = i;
  int c = t % C; t /= C;
  int w = t % W; t /= W;
  int h = t % H;
  int b = t / H;
  int hh = (h - sh) % H; if (hh < 0) hh += H;
  int ww = (w - sw) % W; if (ww < 0) ww += W;
  x[i] += src[(((size_t)b * H + hh) * W + ww) * C + c];
}

// mean pool over spatial positions: f32 [B,HW,C] -> bf16 [B,C]
__global__ __launch_bounds__(256)
void meanpool_kernel(const float* __restrict__ x, __bf16* __restrict__ y,
                     int B, int HW, int C)
{
  int i = blockIdx.x * blockDim.x + threadIdx.x;
  if (i >= B * C) return;
  int b = i / C, c = i % C;
  float s = 0.f;
  for (int t = 0; t < HW; ++t) s += x[((size_t)b * HW + t) * C + c];
  y[i] = f2bf(s / (float)HW);
}

// ==================================================================
// Host orchestration
// ==================================================================
struct LayerP {
  const float *ln1_g, *ln1_b, *qkv_w, *proj_w, *proj_b, *rel_pos;
  const float *ln2_g, *ln2_b, *ff1_w, *ff1_b, *ff2_w, *ff2_b;
};
struct BlockP { const float *ew, *eb, *eg, *ebeta; LayerP L[6]; };

static const int kDepths[4] = {2, 2, 6, 2};
static const int kDims[4]   = {96, 192, 384, 768};
static const int kHeads[4]  = {3, 6, 12, 24};
static const int kFFs[4]    = {384, 768, 1536, 3072};
static const int kReds[4]   = {4, 2, 2, 2};

extern "C" void kernel_launch(void* const* d_in, const int* in_sizes, int n_in,
                              void* d_out, int out_size, void* d_ws, size_t ws_size,
                              hipStream_t stream)
{
  (void)in_sizes; (void)n_in; (void)out_size; (void)ws_size;

  // ---- decode parameter pointers (setup_inputs() dict insertion order) ----
  int idx = 0;
  const float* x_in = (const float*)d_in[idx++];
  BlockP bp[4];
  for (int bi = 0; bi < 4; ++bi) {
    bp[bi].ew    = (const float*)d_in[idx++];
    bp[bi].eb    = (const float*)d_in[idx++];
    bp[bi].eg    = (const float*)d_in[idx++];
    bp[bi].ebeta = (const float*)d_in[idx++];
    for (int li = 0; li < kDepths[bi]; ++li) {
      LayerP& L = bp[bi].L[li];
      L.ln1_g = (const float*)d_in[idx++];  L.ln1_b = (const float*)d_in[idx++];
      L.qkv_w = (const float*)d_in[idx++];
      L.proj_w = (const float*)d_in[idx++]; L.proj_b = (const float*)d_in[idx++];
      L.rel_pos = (const float*)d_in[idx++];
      L.ln2_g = (const float*)d_in[idx++];  L.ln2_b = (const float*)d_in[idx++];
      L.ff1_w = (const float*)d_in[idx++];  L.ff1_b = (const float*)d_in[idx++];
      L.ff2_w = (const float*)d_in[idx++];  L.ff2_b = (const float*)d_in[idx++];
    }
  }
  const float* final_g = (const float*)d_in[idx++];
  const float* final_b = (const float*)d_in[idx++];
  const float* cls_w   = (const float*)d_in[idx++];
  const float* cls_b   = (const float*)d_in[idx++];

  // ---- workspace carve (bytes, 256B aligned) ----
  const size_t ACT = (size_t)64 * 56 * 56 * 96;   // 19,267,584 elems
  const size_t TCH = (size_t)16 * 3136 * 96;      //  4,816,896 elems
  const size_t WMX = (size_t)768 * 3072;          //  2,359,296 elems
  char* base = (char*)d_ws;
  size_t off = 0;
  auto carve = [&](size_t bytes) { void* p = base + off; off += (bytes + 255) & ~(size_t)255; return p; };
  float*   X   = (float*)  carve(ACT * 4);
  __bf16*  T1  = (__bf16*) carve(ACT * 2);
  __bf16*  T2  = (__bf16*) carve(ACT * 2);
  __bf16*  T3  = (__bf16*) carve(TCH * 2);
  float*   T4f = (float*)  carve(TCH * 4);
  __bf16*  T4b = (__bf16*) carve(TCH * 2);
  __bf16*  Wq  = (__bf16*) carve(WMX * 2);
  __bf16*  Wp  = (__bf16*) carve(WMX * 2);
  __bf16*  Wf1 = (__bf16*) carve(WMX * 2);
  __bf16*  Wf2 = (__bf16*) carve(WMX * 2);

  auto gemm = [&](const __bf16* A, const __bf16* BT, const float* bias, const float* resid,
                  void* C, int M, int N, int K, int silu, int obf) {
    dim3 g((unsigned)((M + 127) / 128), (unsigned)((N + 63) / 64));
    gemm_bf16_kernel<<<g, 256, 0, stream>>>(A, BT, bias, resid, C, M, N, K, silu, obf);
  };
  auto twt = [&](const float* W, __bf16* WT, int K, int N) {
    size_t tot = (size_t)K * N;
    transpose_w_kernel<<<(unsigned)((tot + 255) / 256), 256, 0, stream>>>(W, WT, K, N);
  };

  const int Bimg = 64, CB = 16, NCHUNK = Bimg / CB;
  int H = 224, W = 224, C = 3;
  const float* act = x_in;
  int act_nchw = 1;

  for (int bi = 0; bi < 4; ++bi) {
    const int s = kReds[bi], dim = kDims[bi], nh = kHeads[bi], ff = kFFs[bi];
    const int Ho = H / s, Wo = W / s;
    const int M = Bimg * Ho * Wo;
    const int Kg = s * s * C;

    // ---- patch embed: gather(bf16) -> GEMM(+bias, f32 out) -> LN(1e-5)
    {
      size_t total = (size_t)M * Kg;
      patch_gather_kernel<<<(unsigned)((total + 255) / 256), 256, 0, stream>>>(
          act, T1, Bimg, H, W, C, s, act_nchw);
      twt(bp[bi].ew, Wf1, Kg, dim);
      gemm(T1, Wf1, bp[bi].eb, nullptr, X, M, dim, Kg, 0, 0);
      layernorm_kernel<float><<<(M + 7) / 8, 256, 0, stream>>>(
          X, bp[bi].eg, bp[bi].ebeta, X, M, dim, 1e-5f);
    }
    H = Ho; W = Wo; C = dim; act = X; act_nchw = 0;

    const int nwin = (H / 7) * (W / 7);
    const int chunkM = CB * H * W;
    const size_t chunkE = (size_t)chunkM * dim;

    for (int li = 0; li < kDepths[bi]; ++li) {
      const LayerP& L = bp[bi].L[li];
      const int shift = (li % 2 == 0) ? 1 : 0;

      twt(L.qkv_w, Wq, dim, 3 * nh * 32);
      twt(L.proj_w, Wp, nh * 32, dim);
      twt(L.ff1_w, Wf1, dim, ff);
      twt(L.ff2_w, Wf2, ff, dim);

      // ---- attention branch ----
      layernorm_kernel<__bf16><<<(M + 7) / 8, 256, 0, stream>>>(
          X, L.ln1_g, L.ln1_b, T1, M, dim, 1e-6f);
      for (int cc = 0; cc < NCHUNK; ++cc) {
        const __bf16* xs = T1 + (size_t)cc * chunkE;
        const __bf16* qin = xs;
        if (shift) {
          size_t tot8 = chunkE >> 3;
          roll_bf16_kernel<<<(unsigned)((tot8 + 255) / 256), 256, 0, stream>>>(
              xs, T4b, CB, H, W, dim, -3, -3);
          qin = T4b;
        }
        gemm(qin, Wq, nullptr, nullptr, T2, chunkM, 3 * nh * 32, dim, 0, 1);
        win_attn_kernel<<<(unsigned)(CB * nh * nwin), 128, 0, stream>>>(
            T2, L.rel_pos, T3, H, W, nh, shift);
        gemm(T3, Wp, L.proj_b, nullptr, T4f, chunkM, dim, nh * 32, 0, 0);
        roll_add_kernel<<<(unsigned)((chunkE + 255) / 256), 256, 0, stream>>>(
            X + (size_t)cc * chunkE, T4f, CB, H, W, dim,
            shift ? 3 : 0, shift ? 3 : 0);
      }

      // ---- MLP branch ----
      layernorm_kernel<__bf16><<<(M + 7) / 8, 256, 0, stream>>>(
          X, L.ln2_g, L.ln2_b, T1, M, dim, 1e-6f);
      for (int cc = 0; cc < NCHUNK; ++cc) {
        const __bf16* hs = T1 + (size_t)cc * chunkE;
        float* xr = X + (size_t)cc * chunkE;
        gemm(hs, Wf1, L.ff1_b, nullptr, T2, chunkM, ff, dim, 1, 1);
        gemm(T2, Wf2, L.ff2_b, xr, xr, chunkM, dim, ff, 0, 0);
      }
    }
  }

  // ---- head ----
  const int Mf = Bimg * H * W;  // 64 * 49
  layernorm_kernel<float><<<(Mf + 7) / 8, 256, 0, stream>>>(
      X, final_g, final_b, X, Mf, C, 1e-5f);
  meanpool_kernel<<<(Bimg * C + 255) / 256, 256, 0, stream>>>(X, T4b, Bimg, H * W, C);
  twt(cls_w, Wq, C, 1000);
  gemm(T4b, Wq, cls_b, nullptr, (float*)d_out, Bimg, 1000, C, 0, 0);
}
